// Node2Vec_36842229465844
// MI455X (gfx1250) — matrix operand order repaired and checked by reading, not compiled, not run on hardware
//
#include <hip/hip_runtime.h>

typedef __attribute__((ext_vector_type(16))) _Float16 v16h;
typedef __attribute__((ext_vector_type(8)))  float    v8f;

#define EMBED_DIM 64
#define CONTEXT   10
#define POS_ROWS  200000
#define NEG_ROWS  220000
#define POS_G     (POS_ROWS / 16)     // 12500
#define NEG_G     (NEG_ROWS / 16)     // 13750
#define TOT_G     (POS_G + NEG_G)     // 26250
#define EPSF      1e-15f

// Convert 8 contiguous f32 (16B-aligned) to halves v[base..base+8)
__device__ __forceinline__ void cvt8(const float* __restrict__ p, v16h& v, int base) {
    float4 u = *(const float4*)(p);
    float4 w = *(const float4*)(p + 4);
    v[base + 0] = (_Float16)u.x; v[base + 1] = (_Float16)u.y;
    v[base + 2] = (_Float16)u.z; v[base + 3] = (_Float16)u.w;
    v[base + 4] = (_Float16)w.x; v[base + 5] = (_Float16)w.y;
    v[base + 6] = (_Float16)w.z; v[base + 7] = (_Float16)w.w;
}

__global__ void n2v_zero_ws(float* ws) { ws[0] = 0.f; ws[1] = 0.f; }

__global__ void __launch_bounds__(256)
n2v_wmma_kernel(const float* __restrict__ emb,
                const long long* __restrict__ pos_rw,
                const long long* __restrict__ neg_rw,
                float* __restrict__ ws)
{
    const int lane = threadIdx.x & 31;
    const int wid  = (int)((blockIdx.x * blockDim.x + threadIdx.x) >> 5);
    if (wid >= TOT_G) return;                      // wave-uniform; EXEC all-1 below

    const bool isPos = (wid < POS_G);
    const long long* __restrict__ rw = isPos ? pos_rw : neg_rw;
    const int grp = isPos ? wid : (wid - POS_G);
    const int row = grp * 16 + (lane & 15);
    const long long* __restrict__ rowp = rw + (long long)row * CONTEXT;

    // ---- A matrix: start embeddings (16 rows x 64, f32 -> f16) ----
    // 16-bit A 16x32 layout: lane l<16 holds K{0..7,16..23}; lane l+16 holds K{8..15,24..31}
    const int   sidx = (int)rowp[0];
    const float* __restrict__ srow = emb + (long long)sidx * EMBED_DIM;
    const int kA = (lane & 16) ? 8 : 0;
    v16h a0, a1;
    cvt8(srow + kA,       a0, 0);
    cvt8(srow + kA + 16,  a0, 8);
    cvt8(srow + kA + 32,  a1, 0);
    cvt8(srow + kA + 48,  a1, 8);

    // B 32x16 layout: lane n<16 holds K{0..7,16..23} of column n; lane n+16 holds K{8..15,24..31}
    const int kB = (lane & 16) ? 8 : 0;

    // Diagonal D[m,m] lives at: lanes 0-7 -> cc[lane]; lanes 24-31 -> cc[lane-24]
    const int  s   = (lane < 8) ? lane : (lane - 24);
    const bool act = (lane < 8) || (lane >= 24);

    float acc = 0.f;
    #pragma unroll
    for (int c = 0; c < CONTEXT - 1; ++c) {
        const int   ridx = (int)rowp[1 + c];
        const float* __restrict__ rrow = emb + (long long)ridx * EMBED_DIM;
        v16h b0, b1;
        cvt8(rrow + kB,           b0, 0);
        cvt8(rrow + kB + 16,      b0, 8);
        cvt8(rrow + 32 + kB,      b1, 0);
        cvt8(rrow + 32 + kB + 16, b1, 8);

        v8f cc = {};
        cc = __builtin_amdgcn_wmma_f32_16x16x32_f16(false, a0, false, b0,
                                                    (short)0, cc, false, false);
        cc = __builtin_amdgcn_wmma_f32_16x16x32_f16(false, a1, false, b1,
                                                    (short)0, cc, false, false);

        // branchless diagonal extract (3-level select tree)
        float t0 = (s & 1) ? cc[1] : cc[0];
        float t1 = (s & 1) ? cc[3] : cc[2];
        float t2 = (s & 1) ? cc[5] : cc[4];
        float t3 = (s & 1) ? cc[7] : cc[6];
        float u0 = (s & 2) ? t1 : t0;
        float u1 = (s & 2) ? t3 : t2;
        float x  = (s & 4) ? u1 : u0;

        // branchless loss: keep EXEC all-ones for next WMMA iteration
        float sig = 1.0f / (1.0f + __expf(-x));
        float p   = isPos ? (sig + EPSF) : (1.0f - sig + EPSF);
        float li  = -__logf(p);
        acc += act ? li : 0.f;
    }

    // wave32 butterfly reduction
    #pragma unroll
    for (int m = 16; m >= 1; m >>= 1) acc += __shfl_xor(acc, m, 32);
    if (lane == 0) atomicAdd(&ws[isPos ? 0 : 1], acc);
}

__global__ void n2v_finalize(const float* __restrict__ ws, float* __restrict__ out) {
    out[0] = ws[0] / (float)(POS_ROWS * (CONTEXT - 1))
           + ws[1] / (float)(NEG_ROWS * (CONTEXT - 1));
}

extern "C" void kernel_launch(void* const* d_in, const int* in_sizes, int n_in,
                              void* d_out, int out_size, void* d_ws, size_t ws_size,
                              hipStream_t stream) {
    const float*     emb    = (const float*)d_in[0];
    const long long* pos_rw = (const long long*)d_in[1];
    const long long* neg_rw = (const long long*)d_in[2];
    float* ws  = (float*)d_ws;
    float* out = (float*)d_out;

    n2v_zero_ws<<<1, 1, 0, stream>>>(ws);

    const int waves_per_block = 8;                       // 256 threads = 8 wave32
    const int blocks = (TOT_G + waves_per_block - 1) / waves_per_block;
    n2v_wmma_kernel<<<blocks, 256, 0, stream>>>(emb, pos_rw, neg_rw, ws);

    n2v_finalize<<<1, 1, 0, stream>>>(ws, out);
}
